// MultiHeadAttention_55327768707299
// MI455X (gfx1250) — matrix operand order repaired and checked by reading, not compiled, not run on hardware
//
#include <hip/hip_runtime.h>
#include <hip/hip_bf16.h>

// ---- problem constants (match reference) ----
#define BB 4
#define SS 2048
#define DD 512
#define HH 8
#define DK 64
#define HEADSZ ((size_t)SS * DK)          // per (b,h) head plane
#define NPROJ ((size_t)BB * HH * SS * DK) // 4,194,304 floats per q/k/v buffer

typedef __attribute__((ext_vector_type(16))) _Float16 v16h;
typedef __attribute__((ext_vector_type(8)))  float    v8f;

// ---------------------------------------------------------------------------
// WMMA f32 = f16 A(16x32) * f16 B(32x16) + f32 C(16x16)   (wave32, gfx1250)
// ---------------------------------------------------------------------------
__device__ __forceinline__ v8f wmma16x16x32(const v16h a, const v16h b, const v8f c) {
  // 8 args: (neg_a, A, neg_b, B, c_mod, C, reuse_a, reuse_b)
  return __builtin_amdgcn_wmma_f32_16x16x32_f16(false, a, false, b, (short)0, c,
                                                false, false);
}

// A fragment (16x32, f16), ISA 7.12.2: lanes 0-15 -> M=lane, K = {0..7,16..23};
// lanes 16-31 -> M=lane-16, K = {8..15,24..31}. Element p + m*rs + k*cs (fp32->f16).
__device__ __forceinline__ v16h load_a_frag_g(const float* __restrict__ p,
                                              int rs, int cs, int lane) {
  const int m  = lane & 15;
  const int kb = (lane < 16) ? 0 : 8;
  const float* row = p + (size_t)m * rs;
  v16h a;
#pragma unroll
  for (int i = 0; i < 8; ++i) {
    a[i]     = (_Float16)row[(size_t)(kb + i) * cs];
    a[8 + i] = (_Float16)row[(size_t)(kb + 16 + i) * cs];
  }
  return a;
}

// Same layout, but A lives in LDS as f16 already (contiguous K, ld = row stride).
__device__ __forceinline__ v16h load_a_frag_lds(const _Float16* p, int ld, int lane) {
  const int m  = lane & 15;
  const int kb = (lane < 16) ? 0 : 8;
  const _Float16* row = p + (size_t)m * ld;
  v16h a;
#pragma unroll
  for (int i = 0; i < 8; ++i) {
    a[i]     = row[kb + i];
    a[8 + i] = row[kb + 16 + i];
  }
  return a;
}

// B fragment (32x16, f16): lanes 0-15 -> N=lane, K={0..7,16..23}; lanes 16-31 -> N=lane-16,
// K={8..15,24..31}. Element p + k*rs + n*cs (fp32->f16).
__device__ __forceinline__ v16h load_b_frag_g(const float* __restrict__ p,
                                              int rs, int cs, int lane) {
  const int n  = lane & 15;
  const int kb = (lane < 16) ? 0 : 8;
  const float* col = p + (size_t)n * cs;
  v16h b;
#pragma unroll
  for (int i = 0; i < 8; ++i) {
    b[i]     = (_Float16)col[(size_t)(kb + i) * rs];
    b[8 + i] = (_Float16)col[(size_t)(kb + 16 + i) * rs];
  }
  return b;
}

// ---------------------------------------------------------------------------
// Kernel 1: fused linear projection  out[(b*H+h), s, d] = X @ W^T + bias
// grid (M/16 = 512, H = 8), block 128 (4 waves); wave w owns 16 output cols.
// ---------------------------------------------------------------------------
__global__ __launch_bounds__(128) void qkv_proj_kernel(
    const float* __restrict__ X,    // [B*S, 512]
    const float* __restrict__ W,    // [512, 512] (torch Linear: out x in)
    const float* __restrict__ bias, // [512]
    float* __restrict__ outbuf)     // [B*H, S, 64]
{
  const int lane = threadIdx.x & 31;
  const int wave = threadIdx.x >> 5;
  const int m0 = blockIdx.x * 16;            // row in [0, B*S)
  const int h  = blockIdx.y;                 // head (64-col group)
  const int n0 = h * 64 + wave * 16;         // col in [0, 512)

  v8f c = {};
  for (int k0 = 0; k0 < DD; k0 += 32) {
    // speculative prefetch of the A chunk two steps ahead (global_prefetch_b8)
    __builtin_prefetch(X + (size_t)m0 * DD + k0 + 64, 0, 0);
    v16h a = load_a_frag_g(X + (size_t)m0 * DD + k0, DD, 1, lane);
    // B[k][n] = W[n][k]  (W^T)
    v16h b = load_b_frag_g(W + (size_t)n0 * DD + k0, 1, DD, lane);
    c = wmma16x16x32(a, b, c);
  }
  const int col   = lane & 15;
  const int rbase = (lane < 16) ? 0 : 8;
  const float bv  = bias[n0 + col];
  const int bi = m0 / SS;
  const int s0 = m0 % SS;   // 16-row tile never crosses a batch boundary
  float* dst = outbuf + (((size_t)bi * HH + h) * SS + s0) * DK + wave * 16;
#pragma unroll
  for (int r = 0; r < 8; ++r)
    dst[(size_t)(rbase + r) * DK + col] = c[r] + bv;
}

// ---------------------------------------------------------------------------
// Kernel 2: fused attention for one (b,h) x 16-query-row tile.
//   scores = (q k^T)/8 -> mask -> softmax -> attn (written once, fp32)
//   ctx    = attn @ v
// grid (S/16 = 128, B*H = 32), block 128 (4 waves).
// LDS: 16x2048 f16 probability tile (64 KB of the 320 KB WGP LDS).
// ---------------------------------------------------------------------------
__global__ __launch_bounds__(128) void attn_fused_kernel(
    const float* __restrict__ qbuf, const float* __restrict__ kbuf,
    const float* __restrict__ vbuf,
    const unsigned char* __restrict__ mask, // [B,S,S] jax bool (1 byte)
    float* __restrict__ attn_out,           // [B,H,S,S] fp32
    float* __restrict__ ctxbuf)             // [B*H, S, 64]
{
  __shared__ _Float16 Pt[16 * SS];   // 64 KB
  __shared__ float red[4][16];
  __shared__ float rowstat[16];

  const int lane = threadIdx.x & 31;
  const int wave = threadIdx.x >> 5;
  const int bh = blockIdx.y;               // b*H + h
  const int bi = bh / HH;
  const int q0 = blockIdx.x * 16;
  const float* qh = qbuf + (size_t)bh * HEADSZ;
  const float* kh = kbuf + (size_t)bh * HEADSZ;
  const float* vh = vbuf + (size_t)bh * HEADSZ;

  const int col   = lane & 15;
  const int rbase = (lane < 16) ? 0 : 8;

  // ---- Phase 1: scores -> masked/scaled f16 tile in LDS, track row max ----
  // A fragments for the 16 query rows (K = 64 -> two 32-chunks), reused 32x.
  const v16h aq0 = load_a_frag_g(qh + (size_t)q0 * DK + 0,  DK, 1, lane);
  const v16h aq1 = load_a_frag_g(qh + (size_t)q0 * DK + 32, DK, 1, lane);

  float lmax[8];
#pragma unroll
  for (int r = 0; r < 8; ++r) lmax[r] = -3.0e4f;

  const unsigned char* mbase = mask + ((size_t)bi * SS + q0) * SS;
  for (int j = 0; j < 32; ++j) {                 // wave w: cols [w*512, w*512+512)
    const int c0 = wave * 512 + j * 16;
    // B[k][n] = k_head[c0+n][k]  (K^T): k-stride 1, n-stride 64
    v16h b0 = load_b_frag_g(kh + (size_t)c0 * DK + 0,  1, DK, lane);
    v16h b1 = load_b_frag_g(kh + (size_t)c0 * DK + 32, 1, DK, lane);
    v8f c = {};
    c = wmma16x16x32(aq0, b0, c);
    c = wmma16x16x32(aq1, b1, c);
#pragma unroll
    for (int r = 0; r < 8; ++r) {
      float v = c[r] * 0.125f;                   // 1/sqrt(64)
      if (mbase[(size_t)(rbase + r) * SS + c0 + col]) v = -3.0e4f; // f16-safe NEG
      lmax[r] = fmaxf(lmax[r], v);
      Pt[(rbase + r) * SS + c0 + col] = (_Float16)v;
    }
  }
  // reduce max within each 16-lane half (rows rbase..rbase+7)
#pragma unroll
  for (int off = 1; off <= 8; off <<= 1)
#pragma unroll
    for (int r = 0; r < 8; ++r)
      lmax[r] = fmaxf(lmax[r], __shfl_xor(lmax[r], off, 32));
  if (lane == 0 || lane == 16)
#pragma unroll
    for (int r = 0; r < 8; ++r) red[wave][rbase + r] = lmax[r];
  __syncthreads();
  if (threadIdx.x < 16) {
    float m = red[0][threadIdx.x];
#pragma unroll
    for (int w2 = 1; w2 < 4; ++w2) m = fmaxf(m, red[w2][threadIdx.x]);
    rowstat[threadIdx.x] = m;                    // row max
  }
  __syncthreads();

  // ---- exp(x - rowmax) in fp32, partial row sums ----
  for (int r = 0; r < 16; ++r) {
    const float rm = rowstat[r];
    float s = 0.f;
    for (int cc = wave * 512 + lane; cc < wave * 512 + 512; cc += 32) {
      const float e = __expf((float)Pt[r * SS + cc] - rm);
      Pt[r * SS + cc] = (_Float16)e;
      s += e;
    }
#pragma unroll
    for (int off = 1; off < 32; off <<= 1) s += __shfl_xor(s, off, 32);
    if (lane == 0) red[wave][r] = s;
  }
  __syncthreads();
  if (threadIdx.x < 16) {
    const float s = red[0][threadIdx.x] + red[1][threadIdx.x] +
                    red[2][threadIdx.x] + red[3][threadIdx.x];
    rowstat[threadIdx.x] = 1.0f / s;             // 1/denominator
  }
  __syncthreads();

  // ---- normalize, stream attn to HBM (the only touch of the 537MB tensor) ----
  {
    float* arow = attn_out + ((size_t)bh * SS + q0) * SS;
    for (int i = threadIdx.x; i < 16 * SS; i += 128) {
      const float v = (float)Pt[i] * rowstat[i >> 11];
      Pt[i] = (_Float16)v;
      arow[i] = v;                               // coalesced fp32 store
    }
  }
  __syncthreads();

  // ---- Phase 2: ctx(16x64) = P(16x2048) @ V(2048x64); wave w -> 16 cols ----
  v8f acc = {};
  for (int k0 = 0; k0 < SS; k0 += 32) {
    v16h a = load_a_frag_lds(&Pt[k0], SS, lane);
    v16h b = load_b_frag_g(vh + (size_t)k0 * DK + wave * 16, DK, 1, lane);
    acc = wmma16x16x32(a, b, acc);
  }
  float* dst = ctxbuf + ((size_t)bh * SS + q0) * DK + wave * 16;
#pragma unroll
  for (int r = 0; r < 8; ++r)
    dst[(size_t)(rbase + r) * DK + col] = acc[r];
}

// ---------------------------------------------------------------------------
// Kernel 3: out-projection + bias + residual: pre = ctx_flat @ wo^T + bo + Q
// ctx_flat[m][k] lives at ctxbuf[((b*H + k/64)*S + s)*64 + k%64]; each 32-wide
// K-chunk stays inside one head, so it is an affine gather.
// ---------------------------------------------------------------------------
__global__ __launch_bounds__(128) void out_proj_kernel(
    const float* __restrict__ ctxbuf, const float* __restrict__ wo,
    const float* __restrict__ bo, const float* __restrict__ Q,
    float* __restrict__ preout)
{
  const int lane = threadIdx.x & 31;
  const int wave = threadIdx.x >> 5;
  const int m0 = blockIdx.x * 16;
  const int n0 = blockIdx.y * 64 + wave * 16;
  const int bi = m0 >> 11;     // /S
  const int s0 = m0 & (SS - 1);

  v8f c = {};
  for (int k0 = 0; k0 < DD; k0 += 32) {
    const int h = k0 >> 6;
    const float* abase =
        ctxbuf + (((size_t)bi * HH + h) * SS + s0) * DK + (k0 & 63);
    v16h a = load_a_frag_g(abase, DK, 1, lane);
    v16h b = load_b_frag_g(wo + (size_t)n0 * DD + k0, 1, DD, lane);
    c = wmma16x16x32(a, b, c);
  }
  const int col   = lane & 15;
  const int rbase = (lane < 16) ? 0 : 8;
  const float bv  = bo[n0 + col];
#pragma unroll
  for (int r = 0; r < 8; ++r) {
    const size_t row = (size_t)m0 + rbase + r;
    preout[row * DD + n0 + col] = c[r] + bv + Q[row * DD + n0 + col];
  }
}

// ---------------------------------------------------------------------------
// Kernel 4: fp32 LayerNorm over D=512. One 256-thread block per row.
// ---------------------------------------------------------------------------
__global__ __launch_bounds__(256) void layernorm_kernel(
    const float* __restrict__ preout, const float* __restrict__ gamma,
    const float* __restrict__ beta, float* __restrict__ out)
{
  __shared__ float ssum[8], ssq[8];
  const size_t row = blockIdx.x;
  const float* p = preout + row * DD;
  const float x0 = p[threadIdx.x];
  const float x1 = p[threadIdx.x + 256];
  float s = x0 + x1;
  float q = x0 * x0 + x1 * x1;
#pragma unroll
  for (int off = 1; off < 32; off <<= 1) {
    s += __shfl_xor(s, off, 32);
    q += __shfl_xor(q, off, 32);
  }
  const int wave = threadIdx.x >> 5, lane = threadIdx.x & 31;
  if (lane == 0) { ssum[wave] = s; ssq[wave] = q; }
  __syncthreads();
  if (threadIdx.x == 0) {
    float ts = 0.f, tq = 0.f;
#pragma unroll
    for (int i = 0; i < 8; ++i) { ts += ssum[i]; tq += ssq[i]; }
    ssum[0] = ts; ssq[0] = tq;
  }
  __syncthreads();
  const float mean = ssum[0] * (1.0f / DD);
  const float var  = ssq[0] * (1.0f / DD) - mean * mean;
  const float inv  = rsqrtf(var + 1e-5f);
  out[row * DD + threadIdx.x]       = (x0 - mean) * inv * gamma[threadIdx.x] + beta[threadIdx.x];
  out[row * DD + threadIdx.x + 256] = (x1 - mean) * inv * gamma[threadIdx.x + 256] + beta[threadIdx.x + 256];
}

// ---------------------------------------------------------------------------
// Launch. Inputs (setup_inputs order): 0 Q, 1 K, 2 V, 3 attn_mask, 4 n_heads,
// 5 wq, 6 bq, 7 wk, 8 bk, 9 wv, 10 bv, 11 wo, 12 bo, 13 gamma, 14 beta.
// d_out = [out (B*S*D) | attn (B*H*S*S)] fp32.
// d_ws  = q/k/v/ctx/pre buffers (5 * 16.8 MB = 84 MB).
// ---------------------------------------------------------------------------
extern "C" void kernel_launch(void* const* d_in, const int* in_sizes, int n_in,
                              void* d_out, int out_size, void* d_ws, size_t ws_size,
                              hipStream_t stream) {
  (void)in_sizes; (void)n_in; (void)out_size; (void)ws_size;
  const float* Q  = (const float*)d_in[0];
  const float* K  = (const float*)d_in[1];
  const float* V  = (const float*)d_in[2];
  const unsigned char* mask = (const unsigned char*)d_in[3]; // jax bool, 1 byte
  const float* wq = (const float*)d_in[5];
  const float* bq = (const float*)d_in[6];
  const float* wk = (const float*)d_in[7];
  const float* bk = (const float*)d_in[8];
  const float* wv = (const float*)d_in[9];
  const float* bv = (const float*)d_in[10];
  const float* wo = (const float*)d_in[11];
  const float* bo = (const float*)d_in[12];
  const float* gamma = (const float*)d_in[13];
  const float* beta  = (const float*)d_in[14];

  float* out  = (float*)d_out;
  float* attn = out + (size_t)BB * SS * DD;

  float* qbuf = (float*)d_ws;
  float* kbuf = qbuf + NPROJ;
  float* vbuf = kbuf + NPROJ;
  float* ctxb = vbuf + NPROJ;
  float* pre  = ctxb + NPROJ;

  const dim3 blk(128);
  const dim3 gproj(BB * SS / 16, HH);   // (512, 8)
  qkv_proj_kernel<<<gproj, blk, 0, stream>>>(Q, wq, bq, qbuf);
  qkv_proj_kernel<<<gproj, blk, 0, stream>>>(K, wk, bk, kbuf);
  qkv_proj_kernel<<<gproj, blk, 0, stream>>>(V, wv, bv, vbuf);

  attn_fused_kernel<<<dim3(SS / 16, BB * HH), blk, 0, stream>>>(
      qbuf, kbuf, vbuf, mask, attn, ctxb);

  out_proj_kernel<<<dim3(BB * SS / 16, DD / 64), blk, 0, stream>>>(
      ctxb, wo, bo, Q, pre);

  layernorm_kernel<<<dim3(BB * SS), dim3(256), 0, stream>>>(pre, gamma, beta, out);
}
